// AttentionFF_77043123355718
// MI455X (gfx1250) — compile-verified, matrix operand order checked by reading, not run on hardware
//
#include <hip/hip_runtime.h>
#include <math.h>

#define BB 128
#define QQ 384
#define KK 384
#define DM 256
#define HH 8
#define DH 32

typedef __attribute__((ext_vector_type(8)))  float  v8f;
typedef __attribute__((ext_vector_type(8)))  __bf16 v8bf;
typedef __attribute__((ext_vector_type(16))) __bf16 v16bf;

union V16U { v16bf v; v8bf h[2]; };

static __device__ inline v16bf pack16(v8bf lo, v8bf hi) {
  V16U u; u.h[0] = lo; u.h[1] = hi; return u.v;
}

// Per-lane A/B fragment load for V_WMMA_*_16X16X32_BF16.
// lanes 0-15: K = {0..7, 16..23}; lanes 16-31: K = {8..15, 24..31}
// 'row' points at 32 contiguous bf16 K-values (plus k0 base offset).
static __device__ inline v16bf load_bf16_ab(const __bf16* row, int k0, int hi) {
  v8bf lo = *(const v8bf*)(row + k0 + hi * 8);
  v8bf hb = *(const v8bf*)(row + k0 + 16 + hi * 8);
  return pack16(lo, hb);
}

static __device__ inline v16bf load_f32_as_bf16_a(const float* row, int k0, int hi) {
  v8f f0 = *(const v8f*)(row + k0 + hi * 8);
  v8f f1 = *(const v8f*)(row + k0 + 16 + hi * 8);
  V16U u;
#pragma unroll
  for (int i = 0; i < 8; i++) { u.h[0][i] = (__bf16)f0[i]; u.h[1][i] = (__bf16)f1[i]; }
  return u.v;
}

#define WMMA_BF16(a, b, c) \
  __builtin_amdgcn_wmma_f32_16x16x32_bf16(false, (a), false, (b), (short)0, (c), false, false)

// ---------------------------------------------------------------------------
// Kernel 0: weight fp32 -> bf16, transposed so B-operand rows are K-contiguous
// ---------------------------------------------------------------------------
__global__ __launch_bounds__(256) void wprep_kernel(const float* __restrict__ w,
                                                    __bf16* __restrict__ wt) {
  int i = blockIdx.x * blockDim.x + threadIdx.x;  // 0..65535
  int k = i >> 8, n = i & 255;
  wt[(size_t)n * DM + k] = (__bf16)w[(size_t)k * DM + n];
}

// ---------------------------------------------------------------------------
// Kernel 1: fused q/k/v/gate projections. blockIdx.y selects projection.
// Each wave: 16 rows x 64 cols, K-loop of 8 x wmma 16x16x32 per n-tile.
// ---------------------------------------------------------------------------
__global__ __launch_bounds__(128) void proj_kernel(
    const float* __restrict__ qd, const float* __restrict__ md,
    const __bf16* __restrict__ wq, const __bf16* __restrict__ wk,
    const __bf16* __restrict__ wv, const __bf16* __restrict__ wg,
    const float* __restrict__ gate_b,
    __bf16* __restrict__ qo, __bf16* __restrict__ ko,
    __bf16* __restrict__ vto, __bf16* __restrict__ go) {
  const int lane = threadIdx.x & 31;
  const int wid  = threadIdx.x >> 5;
  const int lr = lane & 15, hi = lane >> 4;
  const int m0  = blockIdx.x * 16;
  const int nb0 = wid * 64;
  const int proj = blockIdx.y;

  const float* src = (proj == 0 || proj == 3) ? qd : md;
  const __bf16* wt = (proj == 0) ? wq : (proj == 1) ? wk : (proj == 2) ? wv : wg;

  const float* arow = src + (size_t)(m0 + lr) * DM;

  v8f acc[4] = {};
#pragma unroll
  for (int ks = 0; ks < 8; ks++) {
    int k0 = ks * 32;
    v16bf a = load_f32_as_bf16_a(arow, k0, hi);
#pragma unroll
    for (int t = 0; t < 4; t++) {
      int n = nb0 + t * 16 + lr;
      v16bf bmat = load_bf16_ab(wt + (size_t)n * DM, k0, hi);
      acc[t] = WMMA_BF16(a, bmat, acc[t]);
    }
  }

  const int mrow_add = hi * 8;
#pragma unroll
  for (int t = 0; t < 4; t++) {
    int n = nb0 + t * 16 + lr;
    int h = n >> 5, dh = n & 31;
#pragma unroll
    for (int r = 0; r < 8; r++) {
      int m = m0 + r + mrow_add;
      int bidx = m / QQ, pos = m % QQ;
      float val = acc[t][r];
      if (proj == 0) {  // q: scaled by 1/sqrt(DH), layout [B,H,Q,DH]
        qo[((((size_t)bidx * HH + h) * QQ) + pos) * DH + dh] =
            (__bf16)(val * 0.17677669529663687f);
      } else if (proj == 1) {  // k: [B,H,K,DH]
        ko[((((size_t)bidx * HH + h) * KK) + pos) * DH + dh] = (__bf16)val;
      } else if (proj == 2) {  // v transposed: [B,H,DH,K]
        vto[((((size_t)bidx * HH + h) * DH) + dh) * KK + pos] = (__bf16)val;
      } else {  // gate: sigmoid(x + b), layout [B,Q,256]
        float g = val + gate_b[n];
        go[(size_t)m * DM + n] = (__bf16)(1.0f / (1.0f + __expf(-g)));
      }
    }
  }
}

// ---------------------------------------------------------------------------
// Kernel 2: attention per (b, h, 16 q-rows); single wave per block.
// ---------------------------------------------------------------------------
#define SLP (KK + 8)    // padded f32 logits pitch (392; 392%64==8 -> no bank clash)
#define SPP (KK + 16)   // padded bf16 P pitch (400; 800B rows, 16B aligned)

__global__ __launch_bounds__(32) void attn_kernel(
    const __bf16* __restrict__ qb, const __bf16* __restrict__ kb,
    const __bf16* __restrict__ vt, const __bf16* __restrict__ gate,
    const float* __restrict__ bias, const float* __restrict__ nbb,
    __bf16* __restrict__ wavg) {
  __shared__ float  sl[16 * SLP];
  __shared__ __bf16 sp[16 * SPP];

  const int lane = threadIdx.x;
  const int lr = lane & 15, hi = lane >> 4;
  const int q0 = blockIdx.x * 16;
  const int h  = blockIdx.y;
  const int bidx = blockIdx.z;
  const int mrow_add = hi * 8;

  // Q tile: 16 rows x 32 dh, stays in registers across all k-tiles.
  const __bf16* qbase = qb + ((((size_t)bidx * HH + h) * QQ) + q0) * DH;
  v16bf aq = load_bf16_ab(qbase + (size_t)lr * DH, 0, hi);

  const __bf16* kbase  = kb + (((size_t)bidx * HH + h) * KK) * DH;
  const float*  bias_b = bias + (size_t)bidx * KK;
  const float*  nb_h   = nbb + ((size_t)h * QQ + q0) * KK;

  // logits = q.k^T + bias + nonbatched_bias  -> LDS
  for (int kt = 0; kt < 24; kt++) {
    int kp = kt * 16 + lr;
    v16bf bk = load_bf16_ab(kbase + (size_t)kp * DH, 0, hi);
    v8f c = {};
    c = WMMA_BF16(aq, bk, c);
    float bv = bias_b[kp];
#pragma unroll
    for (int r = 0; r < 8; r++) {
      int q = r + mrow_add;
      sl[q * SLP + kp] = c[r] + bv + nb_h[(size_t)q * KK + kp];
    }
  }
  __syncthreads();

  // softmax over K=384: 2 lanes per row, 192 cols each
  {
    int row = lane >> 1;
    int c0 = (lane & 1) * 192;
    const float* rp = sl + row * SLP + c0;
    float m = -3.4e38f;
    for (int i = 0; i < 192; i++) m = fmaxf(m, rp[i]);
    m = fmaxf(m, __shfl_xor(m, 1));
    float s = 0.f;
    for (int i = 0; i < 192; i++) s += __expf(rp[i] - m);
    s += __shfl_xor(s, 1);
    float inv = 1.0f / s;
    __bf16* pp = sp + row * SPP + c0;
    for (int i = 0; i < 192; i++) pp[i] = (__bf16)(__expf(rp[i] - m) * inv);
  }
  __syncthreads();

  // wavg = P @ V : M=16 q, N=32 dh (two 16-wide tiles), K=384 in 12 steps
  v8f acc0 = {}, acc1 = {};
  const __bf16* vt0 = vt + (((size_t)bidx * HH + h) * DH + lr) * KK;
  const __bf16* vt1 = vt + (((size_t)bidx * HH + h) * DH + 16 + lr) * KK;
#pragma unroll
  for (int ks = 0; ks < 12; ks++) {
    int k0 = ks * 32;
    v16bf ap = load_bf16_ab(sp + lr * SPP, k0, hi);
    v16bf b0 = load_bf16_ab(vt0, k0, hi);
    v16bf b1 = load_bf16_ab(vt1, k0, hi);
    acc0 = WMMA_BF16(ap, b0, acc0);
    acc1 = WMMA_BF16(ap, b1, acc1);
  }

  // gate multiply + store gated wavg as bf16 [B,Q,256]
#pragma unroll
  for (int r = 0; r < 8; r++) {
    int q = q0 + r + mrow_add;
    size_t base = ((size_t)bidx * QQ + q) * DM + h * DH;
    float g0 = (float)gate[base + lr];
    float g1 = (float)gate[base + 16 + lr];
    wavg[base + lr]      = (__bf16)(acc0[r] * g0);
    wavg[base + 16 + lr] = (__bf16)(acc1[r] * g1);
  }
}

// ---------------------------------------------------------------------------
// Kernel 3: output projection [B*Q,256] @ o_w + o_b -> fp32 out
// ---------------------------------------------------------------------------
__global__ __launch_bounds__(128) void oproj_kernel(
    const __bf16* __restrict__ x, const __bf16* __restrict__ wo,
    const float* __restrict__ ob, float* __restrict__ out) {
  const int lane = threadIdx.x & 31;
  const int wid  = threadIdx.x >> 5;
  const int lr = lane & 15, hi = lane >> 4;
  const int m0  = blockIdx.x * 16;
  const int nb0 = wid * 64;

  const __bf16* arow = x + (size_t)(m0 + lr) * DM;

  v8f acc[4] = {};
#pragma unroll
  for (int ks = 0; ks < 8; ks++) {
    int k0 = ks * 32;
    v16bf a = load_bf16_ab(arow, k0, hi);
#pragma unroll
    for (int t = 0; t < 4; t++) {
      int n = nb0 + t * 16 + lr;
      v16bf bmat = load_bf16_ab(wo + (size_t)n * DM, k0, hi);
      acc[t] = WMMA_BF16(a, bmat, acc[t]);
    }
  }

  const int mrow_add = hi * 8;
#pragma unroll
  for (int t = 0; t < 4; t++) {
    int n = nb0 + t * 16 + lr;
    float bn = ob[n];
#pragma unroll
    for (int r = 0; r < 8; r++) {
      int m = m0 + r + mrow_add;
      out[(size_t)m * DM + n] = acc[t][r] + bn;
    }
  }
}

// ---------------------------------------------------------------------------
extern "C" void kernel_launch(void* const* d_in, const int* in_sizes, int n_in,
                              void* d_out, int out_size, void* d_ws, size_t ws_size,
                              hipStream_t stream) {
  (void)in_sizes; (void)n_in; (void)out_size; (void)ws_size;
  const float* q_data = (const float*)d_in[0];
  const float* m_data = (const float*)d_in[1];
  const float* bias   = (const float*)d_in[2];
  const float* nbb    = (const float*)d_in[3];
  const float* q_w    = (const float*)d_in[4];
  const float* k_w    = (const float*)d_in[5];
  const float* v_w    = (const float*)d_in[6];
  const float* o_w    = (const float*)d_in[7];
  const float* o_b    = (const float*)d_in[8];
  const float* g_w    = (const float*)d_in[9];
  const float* g_b    = (const float*)d_in[10];
  float* out = (float*)d_out;

  char* ws = (char*)d_ws;
  size_t off = 0;
  auto alloc = [&](size_t bytes) -> void* {
    void* p = ws + off;
    off = (off + bytes + 255) & ~(size_t)255;
    return p;
  };
  const size_t wbytes = (size_t)DM * DM * sizeof(__bf16);
  const size_t abytes = (size_t)BB * HH * QQ * DH * sizeof(__bf16);
  __bf16* wq_t = (__bf16*)alloc(wbytes);
  __bf16* wk_t = (__bf16*)alloc(wbytes);
  __bf16* wv_t = (__bf16*)alloc(wbytes);
  __bf16* wg_t = (__bf16*)alloc(wbytes);
  __bf16* wo_t = (__bf16*)alloc(wbytes);
  __bf16* qbf  = (__bf16*)alloc(abytes);
  __bf16* kbf  = (__bf16*)alloc(abytes);
  __bf16* vtbf = (__bf16*)alloc(abytes);
  __bf16* gbf  = (__bf16*)alloc((size_t)BB * QQ * DM * sizeof(__bf16));
  __bf16* wavg = (__bf16*)alloc((size_t)BB * QQ * DM * sizeof(__bf16));

  wprep_kernel<<<256, 256, 0, stream>>>(q_w, wq_t);
  wprep_kernel<<<256, 256, 0, stream>>>(k_w, wk_t);
  wprep_kernel<<<256, 256, 0, stream>>>(v_w, wv_t);
  wprep_kernel<<<256, 256, 0, stream>>>(g_w, wg_t);
  wprep_kernel<<<256, 256, 0, stream>>>(o_w, wo_t);

  proj_kernel<<<dim3((BB * QQ) / 16, 4), 128, 0, stream>>>(
      q_data, m_data, wq_t, wk_t, wv_t, wg_t, g_b, qbf, kbf, vtbf, gbf);

  attn_kernel<<<dim3(QQ / 16, HH, BB), 32, 0, stream>>>(
      qbf, kbf, vtbf, gbf, bias, nbb, wavg);

  oproj_kernel<<<(BB * QQ) / 16, 128, 0, stream>>>(wavg, wo_t, o_b, out);
}
